// FP8Linear_46110768890302
// MI455X (gfx1250) — 8-bit matrix operand order repaired and checked by reading, compiled, not run on hardware
//
#include <hip/hip_runtime.h>
#include <stdint.h>


typedef __attribute__((ext_vector_type(16))) int   v16i;
typedef __attribute__((ext_vector_type(8)))  float v8f;
typedef __attribute__((ext_vector_type(4)))  float v4f;
typedef __attribute__((ext_vector_type(4)))  int   v4i;
typedef __attribute__((ext_vector_type(2)))  int   v2i;

#define FP8_MAX 448.0f
#define QBLK 128

#if defined(__AMDGCN__) && __has_builtin(__builtin_amdgcn_global_load_async_to_lds_b128)
#define HAVE_ASYNC_LDS 1
#else
#define HAVE_ASYNC_LDS 0
#endif

#if HAVE_ASYNC_LDS
typedef __attribute__((address_space(1))) v4i gv4i_t;
typedef __attribute__((address_space(3))) v4i lv4i_t;
__device__ __forceinline__ void cp_async16(void* lds, const void* gmem) {
  __builtin_amdgcn_global_load_async_to_lds_b128(
      (gv4i_t*)(unsigned long long)gmem,
      (lv4i_t*)(unsigned)(unsigned long long)lds,
      0, 0);
}
__device__ __forceinline__ void wait_async_le8() {
#if __has_builtin(__builtin_amdgcn_s_wait_asynccnt)
  __builtin_amdgcn_s_wait_asynccnt(8);
#else
  asm volatile("s_wait_asynccnt 8" ::: "memory");
#endif
}
__device__ __forceinline__ void wait_async_le0() {
#if __has_builtin(__builtin_amdgcn_s_wait_asynccnt)
  __builtin_amdgcn_s_wait_asynccnt(0);
#else
  asm volatile("s_wait_asynccnt 0" ::: "memory");
#endif
}
#endif

__device__ __forceinline__ uint32_t f32_to_e4m3_sw(float x) {
  uint32_t u = __float_as_uint(x);
  uint32_t sgn = (u >> 24) & 0x80u;
  float a = fabsf(x);
  if (!(a == a)) return sgn | 0x7Fu;
  if (a >= 448.0f) return sgn | 0x7Eu;
  int e;
  (void)frexpf(a, &e);
  int ex = e - 1;
  if (ex < -6) {
    int q = (int)rintf(a * 512.0f);
    if (q <= 0) return sgn;
    if (q >= 8) return sgn | 0x08u;
    return sgn | (uint32_t)q;
  }
  int q = (int)rintf(ldexpf(a, 3 - ex));
  if (q >= 16) { q >>= 1; ex += 1; }
  if (ex > 8) return sgn | 0x7Eu;
  return sgn | (uint32_t)((ex + 7) << 3) | (uint32_t)(q & 7);
}

__device__ __forceinline__ uint32_t pack4_fp8(float a0, float a1, float a2, float a3) {
#if __has_builtin(__builtin_amdgcn_cvt_pk_fp8_f32)
  int p = 0;
  p = __builtin_amdgcn_cvt_pk_fp8_f32(a0, a1, p, false);
  p = __builtin_amdgcn_cvt_pk_fp8_f32(a2, a3, p, true);
  return (uint32_t)p;
#else
  return f32_to_e4m3_sw(a0) | (f32_to_e4m3_sw(a1) << 8) |
         (f32_to_e4m3_sw(a2) << 16) | (f32_to_e4m3_sw(a3) << 24);
#endif
}

__global__ void __launch_bounds__(256)
fp8_quant_kernel(const float* __restrict__ in, uint32_t* __restrict__ outq,
                 float* __restrict__ scales, int nkb, long long nblk_total) {
  const int wave = threadIdx.x >> 5;
  const int lane = threadIdx.x & 31;
  const long long idx = (long long)blockIdx.x * 8 + wave;
  if (idx >= nblk_total) return;
  const long long r  = idx / nkb;
  const int      kb  = (int)(idx % nkb);
  const long long base = r * (long long)nkb * QBLK + (long long)kb * QBLK;

  v4f v = __builtin_nontemporal_load((const v4f*)(in + base + lane * 4));
  float m = fmaxf(fmaxf(fabsf(v.x), fabsf(v.y)), fmaxf(fabsf(v.z), fabsf(v.w)));
#pragma unroll
  for (int off = 16; off >= 1; off >>= 1)
    m = fmaxf(m, __shfl_xor(m, off, 32));

  const float s   = m / FP8_MAX;
  const float inv = (m > 0.0f) ? (FP8_MAX / m) : 0.0f;
  const uint32_t pk = pack4_fp8(v.x * inv, v.y * inv, v.z * inv, v.w * inv);
  outq[base / 4 + lane] = pk;
  if (lane == 0) scales[r * (long long)nkb + kb] = s;
}

#define BM 128
#define BN 128
#define BK 128
#define LDSP 144

__global__ void __launch_bounds__(256)
fp8_blockscale_gemm(const uint8_t* __restrict__ Aq, const float* __restrict__ As,
                    const uint8_t* __restrict__ Bq, const float* __restrict__ Bs,
                    float* __restrict__ C, int M, int N, int K) {
  __shared__ __align__(16) uint8_t sA[2][BM * LDSP];
  __shared__ __align__(16) uint8_t sB[2][BN * LDSP];
  __shared__ float sXS[BM];
  __shared__ float sWS[BN];

  const int nkb  = K / BK;
  const int tid  = threadIdx.x;
  const int wave = tid >> 5;
  const int lane = tid & 31;
  const int wm = wave >> 2;
  const int wn = wave & 3;
  const int laneLo = lane & 15;
  const int laneHi = lane >> 4;

  const long long m0 = (long long)blockIdx.y * BM;
  const long long n0 = (long long)blockIdx.x * BN;

  const int crow  = tid >> 1;
  const int chalf = (tid & 1) * 64;

  const uint8_t* gAr = Aq + (m0 + crow) * (long long)K + chalf;
  const uint8_t* gBr = Bq + (n0 + crow) * (long long)K + chalf;

  auto issue_copy = [&](int buf, int kb) {
    const uint8_t* gA = gAr + (long long)kb * BK;
    const uint8_t* gB = gBr + (long long)kb * BK;
    uint8_t* lA = &sA[buf][crow * LDSP + chalf];
    uint8_t* lB = &sB[buf][crow * LDSP + chalf];
#if HAVE_ASYNC_LDS
#pragma unroll
    for (int i = 0; i < 4; ++i) cp_async16(lA + 16 * i, gA + 16 * i);
#pragma unroll
    for (int i = 0; i < 4; ++i) cp_async16(lB + 16 * i, gB + 16 * i);
#else
#pragma unroll
    for (int i = 0; i < 4; ++i) {
      *(v4i*)(lA + 16 * i) = *(const v4i*)(gA + 16 * i);
      *(v4i*)(lB + 16 * i) = *(const v4i*)(gB + 16 * i);
    }
#endif
  };

  v8f acc[4][2];
#pragma unroll
  for (int mt = 0; mt < 4; ++mt)
#pragma unroll
    for (int nt = 0; nt < 2; ++nt)
#pragma unroll
      for (int vv = 0; vv < 8; ++vv) acc[mt][nt][vv] = 0.0f;

  issue_copy(0, 0);

  for (int kb = 0; kb < nkb; ++kb) {
    const int cur = kb & 1;
    const bool more = (kb + 1) < nkb;
    if (more) issue_copy(cur ^ 1, kb + 1);

    if (tid < BM) sXS[tid]      = As[(m0 + tid) * nkb + kb];
    else          sWS[tid - BM] = Bs[(n0 + tid - BM) * nkb + kb];

#if HAVE_ASYNC_LDS
    if (more) wait_async_le8();
    else      wait_async_le0();
#endif
    __syncthreads();

    const uint8_t* sAc = sA[cur];
    const uint8_t* sBc = sB[cur];

    v16i bf[2];
    float wsv[2];
#pragma unroll
    for (int nt = 0; nt < 2; ++nt) {
      const int col = wn * 32 + nt * 16 + laneLo;
      const uint8_t* bb = sBc + col * LDSP + laneHi * 16;
#pragma unroll
      for (int j = 0; j < 4; ++j) {
        v4i t = *(const v4i*)(bb + 32 * j);
        bf[nt][4 * j + 0] = t.x; bf[nt][4 * j + 1] = t.y;
        bf[nt][4 * j + 2] = t.z; bf[nt][4 * j + 3] = t.w;
      }
      wsv[nt] = sWS[col];
    }

#pragma unroll
    for (int mt = 0; mt < 4; ++mt) {
      const int row = wm * 64 + mt * 16 + laneLo;
      const uint8_t* ab = sAc + row * LDSP + laneHi * 16;
      v16i af;
#pragma unroll
      for (int j = 0; j < 8; ++j) {
        v2i t = *(const v2i*)(ab + 32 * (j >> 1) + 8 * (j & 1));
        af[2 * j] = t.x; af[2 * j + 1] = t.y;
      }
      const int mrow0 = wm * 64 + mt * 16 + laneHi * 8;
      float xsv[8];
#pragma unroll
      for (int vv = 0; vv < 8; ++vv) xsv[vv] = sXS[mrow0 + vv];

#pragma unroll
      for (int nt = 0; nt < 2; ++nt) {
        v8f zero = {};
        v8f p = __builtin_amdgcn_wmma_f32_16x16x128_fp8_fp8(
            af, bf[nt], (short)0, zero, false, false);
#pragma unroll
        for (int vv = 0; vv < 8; ++vv)
          acc[mt][nt][vv] = fmaf(p[vv], xsv[vv] * wsv[nt], acc[mt][nt][vv]);
      }
    }
    __syncthreads();
  }

#pragma unroll
  for (int mt = 0; mt < 4; ++mt) {
#pragma unroll
    for (int nt = 0; nt < 2; ++nt) {
      const long long col = n0 + wn * 32 + nt * 16 + laneLo;
#pragma unroll
      for (int vv = 0; vv < 8; ++vv) {
        const long long row = m0 + wm * 64 + mt * 16 + laneHi * 8 + vv;
        __builtin_nontemporal_store(acc[mt][nt][vv], &C[row * (long long)N + col]);
      }
    }
  }
}

extern "C" void kernel_launch(void* const* d_in, const int* in_sizes, int n_in,
                              void* d_out, int out_size, void* d_ws, size_t ws_size,
                              hipStream_t stream) {
  const float* x = (const float*)d_in[0];
  const float* w = (const float*)d_in[1];

  const int K = 4096;
  const long long M = (long long)in_sizes[0] / K;
  const long long N = (long long)in_sizes[1] / K;
  const int nkb = K / QBLK;

  uint8_t* base = (uint8_t*)d_ws;
  size_t off = 0;
  uint8_t* xq = base + off; off += (size_t)M * K;          off = (off + 255) & ~(size_t)255;
  uint8_t* wq = base + off; off += (size_t)N * K;          off = (off + 255) & ~(size_t)255;
  float*   xs = (float*)(base + off); off += (size_t)M * nkb * 4; off = (off + 255) & ~(size_t)255;
  float*   wsc = (float*)(base + off);

  const long long xblk = M * nkb;
  const long long wblk = N * nkb;
  fp8_quant_kernel<<<dim3((unsigned)((xblk + 7) / 8)), 256, 0, stream>>>(
      x, (uint32_t*)xq, xs, nkb, xblk);
  fp8_quant_kernel<<<dim3((unsigned)((wblk + 7) / 8)), 256, 0, stream>>>(
      w, (uint32_t*)wq, wsc, nkb, wblk);

  dim3 grid((unsigned)(N / BN), (unsigned)(M / BM));
  fp8_blockscale_gemm<<<grid, 256, 0, stream>>>(
      xq, xs, wq, wsc, (float*)d_out, (int)M, (int)N, K);
}
